// Attention_4148938408732
// MI455X (gfx1250) — matrix operand order repaired and checked by reading, not compiled, not run on hardware
//
#include <hip/hip_runtime.h>

// ---------------------------------------------------------------------------
// MHA forward for MI455X (gfx1250, wave32).
//   - all 4 matmul stages on v_wmma_f32_16x16x32_bf16 (fp32 accumulate)
//   - flash-attention fusion: 1024x1024 score matrices never touch HBM
//   - attention K/V tiles double-buffered in LDS via async global->LDS loads
//     (ASYNCcnt pipeline, CDNA5 ISA)
// ---------------------------------------------------------------------------

#define DIM    512
#define HEADS  8
#define HD     64          // head dim
#define SEQ    1024
#define BATCH  16
#define QSCALE 0.0441941738241592f   // 512^-0.5

typedef __attribute__((ext_vector_type(16))) __bf16 v16bf;
typedef __attribute__((ext_vector_type(8)))  __bf16 v8bf;
typedef __attribute__((ext_vector_type(8)))  float  v8f;

union BF16Frag {
    v16bf v;
    v8bf  h[2];
};

static __device__ __forceinline__ v8f wmma_bf16(v16bf a, v16bf b, v8f c) {
    return __builtin_amdgcn_wmma_f32_16x16x32_bf16(
        /*neg_a=*/false, a, /*neg_b=*/false, b,
        /*c_mod=*/(short)0, c, /*reuse_a=*/false, /*reuse_b=*/false);
}

// ---- async global->LDS (ASYNCcnt path) ------------------------------------
// Builtin parameter types per hipcc diagnostic:
//   arg0: 'vector_size(16) int __device__ *'  (addrspace(1))
//   arg1: LDS-side pointer (addrspace(3)), same vector pointee
typedef int vsi4 __attribute__((vector_size(4 * sizeof(int))));
typedef __attribute__((address_space(1))) vsi4* as1_v4i_p;
typedef __attribute__((address_space(3))) vsi4* as3_v4i_p;

static __device__ __forceinline__ void async_b128(__bf16* lptr, const __bf16* gptr) {
#if __has_builtin(__builtin_amdgcn_global_load_async_to_lds_b128)
    __builtin_amdgcn_global_load_async_to_lds_b128(
        (as1_v4i_p)(unsigned long long)gptr,
        (as3_v4i_p)(unsigned)(unsigned long long)lptr,
        0, 0);
#else
    unsigned           l = (unsigned)(unsigned long long)lptr;
    unsigned long long g = (unsigned long long)gptr;
    asm volatile("global_load_async_to_lds_b128 %0, %1, off"
                 :: "v"(l), "v"(g) : "memory");
#endif
}

#if __has_builtin(__builtin_amdgcn_s_wait_asynccnt)
#define WAIT_ASYNC(n) __builtin_amdgcn_s_wait_asynccnt(n)
#else
#define WAIT_ASYNC(n) asm volatile("s_wait_asynccnt " #n ::: "memory")
#endif

// ---------------------------------------------------------------------------
// Kernel 1: transpose weights to [N][K] bf16 so WMMA B-fragments (lane =
// output column, K contiguous within the lane) are plain b128 loads.
// ---------------------------------------------------------------------------
__global__ void prep_weights(const float* __restrict__ w_qkv,
                             const float* __restrict__ w_out,
                             __bf16* __restrict__ wqkvT,
                             __bf16* __restrict__ woutT) {
    int idx = blockIdx.x * 256 + threadIdx.x;
    if (idx < 3 * DIM * DIM) {
        int n = idx >> 9;
        int k = idx & 511;
        wqkvT[idx] = (__bf16)w_qkv[(size_t)k * (3 * DIM) + n];
    }
    if (idx < DIM * DIM) {
        int n = idx >> 9;
        int k = idx & 511;
        woutT[idx] = (__bf16)w_out[(size_t)k * DIM + n];
    }
}

// ---------------------------------------------------------------------------
// Kernel 2: qkv = x @ w_qkv. One wave per 16x64 output tile (4 accumulators,
// A fragment reused 4x). n0 = blockIdx.x*64 is wave-uniform, and a 64-wide
// tile is exactly one head inside one of Q/K/V -> scalar branches only.
// Writes Q (pre-scaled by 512^-0.5) / K as [b,h,n,d] bf16, V transposed
// [b,h,d,n] bf16 (packed v8bf stores).
// ---------------------------------------------------------------------------
__global__ void qkv_gemm(const float*  __restrict__ x,
                         const __bf16* __restrict__ wqkvT,
                         __bf16* __restrict__ Qs,
                         __bf16* __restrict__ Ks,
                         __bf16* __restrict__ Vt) {
    const int lane = threadIdx.x & 31;
    const int ln   = lane & 15;          // A: row / B: column within 16-tile
    const int g    = lane >> 4;          // lane-half selects K sub-block
    const int row0 = blockIdx.y * 16;    // M tile (b*n flattened)
    const int n0   = blockIdx.x * 64;    // uniform 64-col tile in [0,1536)

    const float*  xrow = x + (size_t)(row0 + ln) * DIM;
    const __bf16* brow[4];
#pragma unroll
    for (int dt = 0; dt < 4; ++dt)
        brow[dt] = wqkvT + (size_t)(n0 + dt * 16 + ln) * DIM;

    v8f zero = {};
    v8f c[4];
#pragma unroll
    for (int dt = 0; dt < 4; ++dt) c[dt] = zero;

    for (int s = 0; s < DIM / 32; ++s) {
        // A fragment: row (ln), K = 32s + {8g..8g+7, 16+8g..16+8g+7}
        const float4* pa = (const float4*)(xrow + 32 * s);
        float4 a0 = pa[2 * g], a1 = pa[2 * g + 1];
        float4 a2 = pa[4 + 2 * g], a3 = pa[5 + 2 * g];
        BF16Frag a;
        a.v[0]=(__bf16)a0.x; a.v[1]=(__bf16)a0.y; a.v[2]=(__bf16)a0.z; a.v[3]=(__bf16)a0.w;
        a.v[4]=(__bf16)a1.x; a.v[5]=(__bf16)a1.y; a.v[6]=(__bf16)a1.z; a.v[7]=(__bf16)a1.w;
        a.v[8]=(__bf16)a2.x; a.v[9]=(__bf16)a2.y; a.v[10]=(__bf16)a2.z; a.v[11]=(__bf16)a2.w;
        a.v[12]=(__bf16)a3.x; a.v[13]=(__bf16)a3.y; a.v[14]=(__bf16)a3.z; a.v[15]=(__bf16)a3.w;
#pragma unroll
        for (int dt = 0; dt < 4; ++dt) {
            BF16Frag b;
            b.h[0] = *(const v8bf*)(brow[dt] + 32 * s + 8 * g);
            b.h[1] = *(const v8bf*)(brow[dt] + 32 * s + 16 + 8 * g);
            c[dt] = wmma_bf16(a.v, b.v, c[dt]);
        }
    }

    // Epilogue: seg/h are wave-uniform (derived from n0 only).
    const int seg   = n0 >> 9;                 // 0=Q 1=K 2=V
    const int h     = (n0 & 511) >> 6;
    const int bb    = row0 >> 10;
    const int nbase = (row0 & 1023) + 8 * g;   // rows v+8g in D layout
    const size_t bh = (size_t)bb * HEADS + h;

    if (seg == 0) {
#pragma unroll
        for (int dt = 0; dt < 4; ++dt) {
            const int d = dt * 16 + ln;
#pragma unroll
            for (int v = 0; v < 8; ++v)
                Qs[(bh * SEQ + nbase + v) * HD + d] = (__bf16)(c[dt][v] * QSCALE);
        }
    } else if (seg == 1) {
#pragma unroll
        for (int dt = 0; dt < 4; ++dt) {
            const int d = dt * 16 + ln;
#pragma unroll
            for (int v = 0; v < 8; ++v)
                Ks[(bh * SEQ + nbase + v) * HD + d] = (__bf16)c[dt][v];
        }
    } else {
#pragma unroll
        for (int dt = 0; dt < 4; ++dt) {
            const int d = dt * 16 + ln;
            v8bf pv;
#pragma unroll
            for (int v = 0; v < 8; ++v) pv[v] = (__bf16)c[dt][v];
            *(v8bf*)(Vt + (bh * HD + d) * SEQ + nbase) = pv;   // 16B packed
        }
    }
}

// ---------------------------------------------------------------------------
// Kernel 3: flash attention, one wave per (b,h, 16-row query tile).
// K/V tiles for each 32-key step are double-buffered in LDS via async
// global->LDS b128 loads (16 per tile); s_wait_asynccnt 16 overlaps the next
// tile's fetch with this tile's 8 WMMAs. Softmax stats live per-row in the
// C/D layout and reduce across the 16-lane halves with shfl_xor; the P tile
// bounces through LDS to re-layout D-fragment -> A-fragment.
// ---------------------------------------------------------------------------
__global__ void attn_kernel(const __bf16* __restrict__ Qs,
                            const __bf16* __restrict__ Ks,
                            const __bf16* __restrict__ Vt,
                            __bf16* __restrict__ attn_out) {
    __shared__ __bf16 lds_p[16 * 32];        // 1 KB P tile
    __shared__ __bf16 kbuf[2][32 * 64];      // 2 x 4 KB K tiles [j][d]
    __shared__ __bf16 vbuf[2][64 * 32];      // 2 x 4 KB V tiles [d][j]

    const int lane = threadIdx.x & 31;
    const int ln   = lane & 15;
    const int g    = lane >> 4;
    const int i0   = blockIdx.x * 16;        // query-row tile
    const int bh   = blockIdx.y;             // b*8 + h
    const int b    = bh >> 3;
    const int h    = bh & 7;

    // Q A-fragments for d = 0..31 and 32..63 (already scaled)
    const __bf16* qrow = Qs + ((size_t)bh * SEQ + i0 + ln) * HD;
    BF16Frag qa[2];
#pragma unroll
    for (int s = 0; s < 2; ++s) {
        qa[s].h[0] = *(const v8bf*)(qrow + 32 * s + 8 * g);
        qa[s].h[1] = *(const v8bf*)(qrow + 32 * s + 16 + 8 * g);
    }

    const __bf16* kbase0 = Ks + (size_t)bh * SEQ * HD;
    const __bf16* vbase0 = Vt + (size_t)bh * HD * SEQ;

    // Prefetch one (K,V) tile: 32x64 + 64x32 bf16 = 2 x 4KB = 16 async b128.
    auto prefetch = [&](int buf, int j0) {
        const __bf16* kb = kbase0 + (size_t)j0 * HD;
        const __bf16* vb = vbase0 + j0;
#pragma unroll
        for (int t = 0; t < 8; ++t) {
            const int idx = t * 32 + lane;               // 0..255 16B segments
            const int kr = idx >> 3, ks = idx & 7;       // K: 32 rows x 8 segs
            async_b128(&kbuf[buf][kr * 64 + ks * 8], kb + (size_t)kr * HD + ks * 8);
            const int vr = idx >> 2, vs = idx & 3;       // V: 64 rows x 4 segs
            async_b128(&vbuf[buf][vr * 32 + vs * 8], vb + (size_t)vr * SEQ + vs * 8);
        }
    };

    v8f zero = {};
    v8f acc[4];
#pragma unroll
    for (int t = 0; t < 4; ++t) acc[t] = zero;
    float mrow[8], lrow[8];
#pragma unroll
    for (int v = 0; v < 8; ++v) { mrow[v] = -3.0e38f; lrow[v] = 0.0f; }

    prefetch(0, 0);

    for (int j0 = 0; j0 < SEQ; j0 += 32) {
        const int buf = (j0 >> 5) & 1;
        if (j0 + 32 < SEQ) {
            prefetch(buf ^ 1, j0 + 32);   // 16 newest in flight = next tile
            WAIT_ASYNC(16);               // current tile resident in LDS
        } else {
            WAIT_ASYNC(0);
        }

        // ---- scores: S[16 x 32] = Q @ K^T (two 16-col tiles, K from LDS) ----
        v8f s0 = zero, s1 = zero;
#pragma unroll
        for (int s = 0; s < 2; ++s) {
            const __bf16* k0 = &kbuf[buf][(ln)      * 64 + 32 * s + 8 * g];
            const __bf16* k1 = &kbuf[buf][(16 + ln) * 64 + 32 * s + 8 * g];
            BF16Frag kb0, kb1;
            kb0.h[0] = *(const v8bf*)k0;  kb0.h[1] = *(const v8bf*)(k0 + 16);
            kb1.h[0] = *(const v8bf*)k1;  kb1.h[1] = *(const v8bf*)(k1 + 16);
            s0 = wmma_bf16(qa[s].v, kb0.v, s0);
            s1 = wmma_bf16(qa[s].v, kb1.v, s1);
        }

        // ---- online softmax (per row v; stats replicated across the half) ----
#pragma unroll
        for (int v = 0; v < 8; ++v) {
            float mv = fmaxf(s0[v], s1[v]);
#pragma unroll
            for (int off = 1; off < 16; off <<= 1)
                mv = fmaxf(mv, __shfl_xor(mv, off, 32));
            const float mnew  = fmaxf(mrow[v], mv);
            const float alpha = __expf(mrow[v] - mnew);
            const float p0 = __expf(s0[v] - mnew);
            const float p1 = __expf(s1[v] - mnew);
            float rs = p0 + p1;
#pragma unroll
            for (int off = 1; off < 16; off <<= 1)
                rs += __shfl_xor(rs, off, 32);
            lrow[v] = lrow[v] * alpha + rs;
            mrow[v] = mnew;
            acc[0][v] *= alpha; acc[1][v] *= alpha;
            acc[2][v] *= alpha; acc[3][v] *= alpha;
            const int r = v + 8 * g;
            lds_p[r * 32 + ln]      = (__bf16)p0;
            lds_p[r * 32 + 16 + ln] = (__bf16)p1;
        }
        __syncthreads();                     // D-layout -> A-layout bounce

        BF16Frag pa;                         // P as A operand (K = 32 keys)
        pa.h[0] = *(const v8bf*)(&lds_p[ln * 32 + 8 * g]);
        pa.h[1] = *(const v8bf*)(&lds_p[ln * 32 + 16 + 8 * g]);

        // ---- acc += P @ V (V tile from LDS, row = d, contiguous j) ----
#pragma unroll
        for (int dt = 0; dt < 4; ++dt) {
            const __bf16* vrow = &vbuf[buf][(dt * 16 + ln) * 32 + 8 * g];
            BF16Frag vb;
            vb.h[0] = *(const v8bf*)vrow;
            vb.h[1] = *(const v8bf*)(vrow + 16);
            acc[dt] = wmma_bf16(pa.v, vb.v, acc[dt]);
        }
        __syncthreads();
    }

    // ---- normalize and write to [b, n, h*64 + d] bf16 ----
#pragma unroll
    for (int v = 0; v < 8; ++v) {
        const float inv = 1.0f / lrow[v];
        const int row = i0 + v + 8 * g;
#pragma unroll
        for (int dt = 0; dt < 4; ++dt) {
            attn_out[((size_t)b * SEQ + row) * DIM + h * HD + dt * 16 + ln] =
                (__bf16)(acc[dt][v] * inv);
        }
    }
}

// ---------------------------------------------------------------------------
// Kernel 4: out = attn_out @ w_out + b_out (fp32). One wave per 16x64 tile.
// ---------------------------------------------------------------------------
__global__ void out_gemm(const __bf16* __restrict__ attn_out,
                         const __bf16* __restrict__ woutT,
                         const float*  __restrict__ b_out,
                         float* __restrict__ out) {
    const int lane = threadIdx.x & 31;
    const int ln   = lane & 15;
    const int g    = lane >> 4;
    const int row0 = blockIdx.y * 16;
    const int n0   = blockIdx.x * 64;

    const __bf16* arow = attn_out + (size_t)(row0 + ln) * DIM;
    const __bf16* brow[4];
#pragma unroll
    for (int dt = 0; dt < 4; ++dt)
        brow[dt] = woutT + (size_t)(n0 + dt * 16 + ln) * DIM;

    v8f zero = {};
    v8f c[4];
#pragma unroll
    for (int dt = 0; dt < 4; ++dt) c[dt] = zero;

    for (int s = 0; s < DIM / 32; ++s) {
        BF16Frag a;
        a.h[0] = *(const v8bf*)(arow + 32 * s + 8 * g);
        a.h[1] = *(const v8bf*)(arow + 32 * s + 16 + 8 * g);
#pragma unroll
        for (int dt = 0; dt < 4; ++dt) {
            BF16Frag bfr;
            bfr.h[0] = *(const v8bf*)(brow[dt] + 32 * s + 8 * g);
            bfr.h[1] = *(const v8bf*)(brow[dt] + 32 * s + 16 + 8 * g);
            c[dt] = wmma_bf16(a.v, bfr.v, c[dt]);
        }
    }

#pragma unroll
    for (int dt = 0; dt < 4; ++dt) {
        const int cidx = n0 + dt * 16 + ln;
        const float bias = b_out[cidx];
#pragma unroll
        for (int v = 0; v < 8; ++v) {
            const int row = row0 + v + 8 * g;
            out[(size_t)row * DIM + cidx] = c[dt][v] + bias;
        }
    }
}

// ---------------------------------------------------------------------------
extern "C" void kernel_launch(void* const* d_in, const int* in_sizes, int n_in,
                              void* d_out, int out_size, void* d_ws, size_t ws_size,
                              hipStream_t stream) {
    (void)in_sizes; (void)n_in; (void)out_size; (void)ws_size;
    const float* x     = (const float*)d_in[0];
    const float* w_qkv = (const float*)d_in[1];
    const float* w_out = (const float*)d_in[2];
    const float* b_out = (const float*)d_in[3];
    float* out = (float*)d_out;

    // Workspace layout (all bf16), ~69 MB total.
    char* ws = (char*)d_ws;
    __bf16* wqkvT = (__bf16*)ws;  ws += (size_t)3 * DIM * DIM * 2;   // [1536][512]
    __bf16* woutT = (__bf16*)ws;  ws += (size_t)DIM * DIM * 2;       // [512][512]
    const size_t qkv_elems = (size_t)BATCH * HEADS * SEQ * HD;
    __bf16* Qs = (__bf16*)ws;     ws += qkv_elems * 2;               // [b,h,n,d]
    __bf16* Ks = (__bf16*)ws;     ws += qkv_elems * 2;               // [b,h,n,d]
    __bf16* Vt = (__bf16*)ws;     ws += qkv_elems * 2;               // [b,h,d,n]
    __bf16* attn_o = (__bf16*)ws;                                    // [b,n,512]

    prep_weights<<<3072, 256, 0, stream>>>(w_qkv, w_out, wqkvT, woutT);

    // M=16384 -> 1024 row tiles; N=1536 -> 24 uniform 64-col tiles.
    qkv_gemm<<<dim3(24, 1024), 32, 0, stream>>>(x, wqkvT, Qs, Ks, Vt);

    // 64 query tiles x 128 (b,h) pairs, one wave each.
    attn_kernel<<<dim3(64, 128), 32, 0, stream>>>(Qs, Ks, Vt, attn_o);

    // N=512 -> 8 uniform 64-col tiles.
    out_gemm<<<dim3(8, 1024), 32, 0, stream>>>(attn_o, woutT, b_out, out);
}
